// WMDSimilarityLoss_13280038879663
// MI455X (gfx1250) — matrix operand order repaired and checked by reading, not compiled, not run on hardware
//
#include <hip/hip_runtime.h>

typedef __attribute__((ext_vector_type(16))) _Float16 v16h;
typedef __attribute__((ext_vector_type(8)))  float    v8f;

#define BB   64
#define MM   512
#define NN   512
#define DD   128
#define EPSV 0.05f
#define NIT  200
#define LOGP (-6.2383246250f)   // -log(512)

// ---------------------------------------------------------------------------
// Phase A: D[b,i,j] = sqrt(max((|x_i|^2 + |y_j|^2 - 2 x_i.y_j)/128, 1e-12))
// One wave per 16x16 output tile. x.y^T via V_WMMA_F32_16X16X32_F16 with
// hi/lo fp16 split (3 WMMAs per K-block) for ~fp32 accuracy.
// ---------------------------------------------------------------------------
__global__ __launch_bounds__(256) void wmd_cost_kernel(
    const float* __restrict__ x, const float* __restrict__ y,
    float* __restrict__ Dmat) {
  const int lane = threadIdx.x & 31;
  const int wave = threadIdx.x >> 5;
  const int tile = blockIdx.x * 8 + wave;          // 64*32*32 tiles total
  const int b  = tile >> 10;
  const int ti = (tile >> 5) & 31;
  const int tj = tile & 31;
  const int i0 = ti * 16, j0 = tj * 16;
  const int half = lane >> 4;                      // K-half selector
  const int lrow = lane & 15;

  const float* xa = x + ((size_t)b * MM + (i0 + lrow)) * DD;
  const float* yb = y + ((size_t)b * NN + (j0 + lrow)) * DD;

  v8f acc = {};
  float xpart = 0.f, ypart = 0.f;

#pragma unroll
  for (int kb = 0; kb < 4; ++kb) {
    // A fragment (16x32 f16 layout): lanes 0-15 K={0..7,16..23}, lanes 16-31 +8
    const int ka = kb * 32 + half * 8;
    alignas(16) float af[16];
    *(float4*)(af + 0)  = *(const float4*)(xa + ka + 0);
    *(float4*)(af + 4)  = *(const float4*)(xa + ka + 4);
    *(float4*)(af + 8)  = *(const float4*)(xa + ka + 16);
    *(float4*)(af + 12) = *(const float4*)(xa + ka + 20);
    // B fragment (32x16 f16 layout): lane's column = j0+lrow, K={0..7,16..23}, lanes 16-31 +8
    const int kc = kb * 32 + half * 8;
    alignas(16) float bf[16];
    *(float4*)(bf + 0)  = *(const float4*)(yb + kc + 0);
    *(float4*)(bf + 4)  = *(const float4*)(yb + kc + 4);
    *(float4*)(bf + 8)  = *(const float4*)(yb + kc + 16);
    *(float4*)(bf + 12) = *(const float4*)(yb + kc + 20);

    v16h ah, al, bh, bl;
#pragma unroll
    for (int e = 0; e < 16; ++e) {
      float a = af[e];
      _Float16 h = (_Float16)a;
      ah[e] = h; al[e] = (_Float16)(a - (float)h);
      xpart += a * a;
      float bv = bf[e];
      _Float16 hb = (_Float16)bv;
      bh[e] = hb; bl[e] = (_Float16)(bv - (float)hb);
      ypart += bv * bv;
    }
    // hi*hi + hi*lo + lo*hi  (lo*lo ~ 2^-44, dropped)
    acc = __builtin_amdgcn_wmma_f32_16x16x32_f16(false, ah, false, bh,
                                                 (short)0, acc, false, false);
    acc = __builtin_amdgcn_wmma_f32_16x16x32_f16(false, ah, false, bl,
                                                 (short)0, acc, false, false);
    acc = __builtin_amdgcn_wmma_f32_16x16x32_f16(false, al, false, bh,
                                                 (short)0, acc, false, false);
  }
  // complete row sums of squares: other half of K lives in lane^16
  float xn = xpart + __shfl_xor(xpart, 16);
  float yn = ypart + __shfl_xor(ypart, 16);   // lane holds yn of its own column

  float* Dout = Dmat + (size_t)b * MM * NN;
#pragma unroll
  for (int r = 0; r < 8; ++r) {
    const int row = r + half * 8;             // C/D layout: VGPR r, lane half
    float xr = __shfl(xn, row);               // xn[m] lives in lane m
    float sq = (xr + yn - 2.0f * acc[r]) * (1.0f / 128.0f);
    Dout[(size_t)(i0 + row) * NN + (j0 + lrow)] = sqrtf(fmaxf(sq, 1e-12f));
  }
}

// ---------------------------------------------------------------------------
// Phase B: log-domain Sinkhorn, one workgroup (512 thr = 16 waves) per batch.
// f,g live in LDS; D streams from L2 (64MB working set < 192MB L2).
// ---------------------------------------------------------------------------
__global__ __launch_bounds__(512) void wmd_sinkhorn_kernel(
    const float* __restrict__ Dmat, float* __restrict__ wmd) {
  const int b    = blockIdx.x;
  const int tid  = threadIdx.x;
  const int lane = tid & 31;
  const int wave = tid >> 5;
  __shared__ __align__(16) float fs[MM];
  __shared__ __align__(16) float gs[NN];
  __shared__ float red[16];

  const float* Db = Dmat + (size_t)b * MM * NN;
  const float inv_eps = 1.0f / EPSV;
  fs[tid] = 0.f; gs[tid] = 0.f;
  __syncthreads();

  for (int it = 0; it < NIT; ++it) {
    // ---- f update: wave handles rows wave, wave+16, ... (coalesced j)
    for (int row = wave; row < MM; row += 16) {
      const float* Dr = Db + (size_t)row * NN;
      if (row + 16 < MM)   // prefetch next row this wave will touch
        __builtin_prefetch(Db + (size_t)(row + 16) * NN + lane * 16, 0, 0);
      float m = -3.0e38f, s = 0.f;
#pragma unroll
      for (int c = 0; c < 4; ++c) {
        const int j = c * 128 + lane * 4;
        float4 d4 = *(const float4*)(Dr + j);
        float4 g4 = *(const float4*)(&gs[j]);
        float v[4] = {(g4.x - d4.x) * inv_eps, (g4.y - d4.y) * inv_eps,
                      (g4.z - d4.z) * inv_eps, (g4.w - d4.w) * inv_eps};
#pragma unroll
        for (int k = 0; k < 4; ++k) {
          float nm = fmaxf(m, v[k]);
          s = s * __expf(m - nm) + __expf(v[k] - nm);
          m = nm;
        }
      }
#pragma unroll
      for (int off = 16; off > 0; off >>= 1) {   // merge (m,s) across wave32
        float om = __shfl_xor(m, off), os = __shfl_xor(s, off);
        float nm = fmaxf(m, om);
        s = s * __expf(m - nm) + os * __expf(om - nm);
        m = nm;
      }
      if (lane == 0) fs[row] = EPSV * (LOGP - (m + __logf(s)));
    }
    __syncthreads();
    // ---- g update: thread owns column tid (block reads 512 contiguous/step)
    {
      const float* Dc = Db + tid;
      float m = -3.0e38f, s = 0.f;
      for (int i = 0; i < MM; ++i) {
        float v = (fs[i] - Dc[(size_t)i * NN]) * inv_eps;
        float nm = fmaxf(m, v);
        s = s * __expf(m - nm) + __expf(v - nm);
        m = nm;
      }
      gs[tid] = EPSV * (LOGP - (m + __logf(s)));
    }
    __syncthreads();
  }

  // ---- WMD = sum_ij exp((f_i+g_j-D_ij)/eps) * D_ij
  float part = 0.f;
  {
    const float* Dc = Db + tid;
    const float gc = gs[tid];
    for (int i = 0; i < MM; ++i) {
      float d = Dc[(size_t)i * NN];
      part += __expf((fs[i] + gc - d) * inv_eps) * d;
    }
  }
#pragma unroll
  for (int off = 16; off > 0; off >>= 1) part += __shfl_xor(part, off);
  if (lane == 0) red[wave] = part;
  __syncthreads();
  if (wave == 0) {
    float v = (lane < 16) ? red[lane] : 0.f;
#pragma unroll
    for (int off = 8; off > 0; off >>= 1) v += __shfl_xor(v, off);
    if (lane == 0) wmd[b] = v;
  }
}

// ---------------------------------------------------------------------------
// Phase C: loss = mean((wmd - labels)^2)
// ---------------------------------------------------------------------------
__global__ void wmd_mse_kernel(const float* __restrict__ wmd,
                               const float* __restrict__ labels,
                               float* __restrict__ out) {
  __shared__ float red[64];
  const int t = threadIdx.x;
  float e = wmd[t] - labels[t];
  red[t] = e * e;
  __syncthreads();
  for (int off = 32; off > 0; off >>= 1) {
    if (t < off) red[t] += red[t + off];
    __syncthreads();
  }
  if (t == 0) out[0] = red[0] * (1.0f / 64.0f);
}

extern "C" void kernel_launch(void* const* d_in, const int* in_sizes, int n_in,
                              void* d_out, int out_size, void* d_ws,
                              size_t ws_size, hipStream_t stream) {
  const float* x      = (const float*)d_in[0];
  const float* y      = (const float*)d_in[1];
  const float* labels = (const float*)d_in[2];
  float* Dmat = (float*)d_ws;                            // 64 MiB
  float* wmd  = (float*)d_ws + (size_t)BB * MM * NN;     // 64 floats

  // 64*32*32 = 65536 tiles, 8 waves (tiles) per 256-thread block
  wmd_cost_kernel<<<(BB * 32 * 32) / 8, 256, 0, stream>>>(x, y, Dmat);
  wmd_sinkhorn_kernel<<<BB, 512, 0, stream>>>(Dmat, wmd);
  wmd_mse_kernel<<<1, 64, 0, stream>>>(wmd, labels, (float*)d_out);
}
